// CNN_BiMAMBA_AV_10746008174986
// MI455X (gfx1250) — matrix operand order repaired and checked by reading, compiled, not run on hardware
//
#include <hip/hip_runtime.h>
#include <stdint.h>

// ---------------------------------------------------------------------------
// Types for CDNA5 WMMA (wave32): 16x16x32 bf16 -> f32 accumulate
// ---------------------------------------------------------------------------
typedef __attribute__((ext_vector_type(16))) __bf16 v16bf;
typedef __attribute__((ext_vector_type(8)))  float  v8f;

#define WAVE 32

__device__ __forceinline__ unsigned short f32_to_bf16(float f) {
    union { float f; unsigned int u; } v; v.f = f;
    unsigned int u = v.u;
    u += 0x7FFFu + ((u >> 16) & 1u);      // round-to-nearest-even
    return (unsigned short)(u >> 16);
}

// A-fragment (16x32, 16-bit): lane L holds row M=L%16, K = {c..c+7} U {c+16..c+23}, c=(L/16)*8
__device__ __forceinline__ v16bf frag_load_a(const unsigned short* p) {
    union { uint4 u[2]; v16bf v; } u;
    u.u[0] = *(const uint4*)(p);
    u.u[1] = *(const uint4*)(p + 16);
    return u.v;
}
// B-fragment (32x16, 16-bit): lane L holds col N=L%16, K = {c..c+7} U {c+16..c+23}, c=(L/16)*8
__device__ __forceinline__ v16bf frag_load_b(const unsigned short* p) {
    union { uint4 u[2]; v16bf v; } u;
    u.u[0] = *(const uint4*)(p);
    u.u[1] = *(const uint4*)(p + 16);
    return u.v;
}
__device__ __forceinline__ v16bf frag_zero() {
    union { unsigned int u[8]; v16bf v; } z;
#pragma unroll
    for (int i = 0; i < 8; ++i) z.u[i] = 0u;
    return z.v;
}
__device__ __forceinline__ v8f wmma_bf16(v16bf a, v16bf b, v8f c) {
    return __builtin_amdgcn_wmma_f32_16x16x32_bf16(false, a, false, b, (short)0, c, false, false);
}

// ---------------------------------------------------------------------------
// Model constants
// ---------------------------------------------------------------------------
#define BSZ   8
#define LSEQ  128
#define NIMG  (BSZ * LSEQ)          // 1024
#define NMFCC 60
#define VFEAT 196
#define DMODEL 256
#define EDIM   512                  // 2*D
#define NST    16
#define DTR    16
#define NLAYER 4

// ---------------------------------------------------------------------------
// Utility kernels
// ---------------------------------------------------------------------------
__global__ void convert_bf16_kernel(const float* __restrict__ src,
                                    unsigned short* __restrict__ dst, int n) {
    int i = blockIdx.x * 256 + threadIdx.x;
    if (i < n) dst[i] = f32_to_bf16(src[i]);
}

__global__ void zero_f32_kernel(float* __restrict__ p, int n) {
    int i = blockIdx.x * 256 + threadIdx.x;
    if (i < n) p[i] = 0.0f;
}

// ---------------------------------------------------------------------------
// conv1: 3->32, 3x3 SAME on 64x64, relu, maxpool2 -> bf16 [img][32][32][32ch]
// ---------------------------------------------------------------------------
__global__ __launch_bounds__(256)
void conv1_kernel(const float* __restrict__ vid,   // [1024][3][64][64]
                  const float* __restrict__ w1,    // [32][3][3][3]
                  const float* __restrict__ b1,    // [32]
                  unsigned short* __restrict__ out) {
    int id = blockIdx.x * 256 + threadIdx.x;       // 1024*32*32*32
    int c   = id & 31;
    int x   = (id >> 5) & 31;
    int y   = (id >> 10) & 31;
    int img = id >> 15;
    if (img >= NIMG) return;

    float wreg[27];
#pragma unroll
    for (int i = 0; i < 27; ++i) wreg[i] = w1[c * 27 + i];
    float bias = b1[c];

    float best = 0.0f;                              // relu >= 0
#pragma unroll
    for (int dy = 0; dy < 2; ++dy)
#pragma unroll
    for (int dx = 0; dx < 2; ++dx) {
        int Y = 2 * y + dy, X = 2 * x + dx;
        float s = bias;
#pragma unroll
        for (int ic = 0; ic < 3; ++ic)
#pragma unroll
        for (int ky = 0; ky < 3; ++ky) {
            int yy = Y + ky - 1;
            if (yy < 0 || yy >= 64) continue;
#pragma unroll
            for (int kx = 0; kx < 3; ++kx) {
                int xx = X + kx - 1;
                if (xx < 0 || xx >= 64) continue;
                s += vid[(((size_t)img * 3 + ic) * 64 + yy) * 64 + xx] *
                     wreg[(ic * 3 + ky) * 3 + kx];
            }
        }
        best = s > best ? s : best;
    }
    out[id] = f32_to_bf16(best);
}

// pack cnn_conv2_w (64,32,3,3) OIHW f32 -> [tap(9)][oc(64)][ic(32)] bf16
__global__ void packw2_kernel(const float* __restrict__ w2,
                              unsigned short* __restrict__ out) {
    int i = blockIdx.x * 256 + threadIdx.x;
    if (i >= 9 * 64 * 32) return;
    int ic = i & 31, oc = (i >> 5) & 63, tap = i >> 11;
    out[i] = f32_to_bf16(w2[((size_t)oc * 32 + ic) * 9 + tap]);
}

// ---------------------------------------------------------------------------
// conv2 as implicit WMMA GEMM + fused bias/relu/maxpool/mean
// M-tile = 4 pools x 4 corners (so rows j0..3 / j4..7 of C are whole pools),
// N = 64 out channels (4 WMMA tiles per wave), K-loop = 9 taps x 32 in-ch.
// Accumulates the spatial mean into feat[img][oc] via atomics.
// ---------------------------------------------------------------------------
__global__ __launch_bounds__(256)
void conv2_wmma_kernel(const unsigned short* __restrict__ c1,   // [1024][32][32][32]
                       const unsigned short* __restrict__ w2,   // [9][64][32] bf16
                       const float* __restrict__ b2,            // [64]
                       float* __restrict__ feat) {              // [1024][64]
    int wave = blockIdx.x * 8 + (threadIdx.x >> 5);
    int lane = threadIdx.x & 31;
    int img = wave >> 6;
    int t   = wave & 63;
    int ty  = t >> 2, tx = t & 3;

    int m = lane & 15;
    int q = m >> 2, corner = m & 3;
    int y = 2 * ty + (corner >> 1);
    int x = 2 * (tx * 4 + q) + (corner & 1);
    int kh = (lane >> 4) * 8;

    v8f acc[4];
#pragma unroll
    for (int n = 0; n < 4; ++n) acc[n] = v8f{0, 0, 0, 0, 0, 0, 0, 0};

#pragma unroll
    for (int tap = 0; tap < 9; ++tap) {
        int ky = tap / 3, kx = tap % 3;
        int yy = y + ky - 1, xx = x + kx - 1;
        v16bf a;
        if (yy >= 0 && yy < 32 && xx >= 0 && xx < 32) {
            const unsigned short* p =
                c1 + (((size_t)img * 32 + yy) * 32 + xx) * 32 + kh;
            a = frag_load_a(p);
        } else {
            a = frag_zero();
        }
        const unsigned short* wbase = w2 + (size_t)tap * 64 * 32;
#pragma unroll
        for (int nt = 0; nt < 4; ++nt) {
            const unsigned short* wp =
                wbase + ((size_t)(nt * 16 + (lane & 15))) * 32 + (lane >> 4) * 8;
            v16bf b = frag_load_b(wp);
            acc[nt] = wmma_bf16(a, b, acc[nt]);
        }
    }

#pragma unroll
    for (int nt = 0; nt < 4; ++nt) {
        int oc = nt * 16 + (lane & 15);
        float bv = b2[oc];
        float p0 = 0.0f, p1 = 0.0f;               // relu'd values >= 0
#pragma unroll
        for (int j = 0; j < 8; ++j) {
            float v = acc[nt][j] + bv;
            v = v > 0.0f ? v : 0.0f;
            if (j < 4) p0 = v > p0 ? v : p0;
            else       p1 = v > p1 ? v : p1;
        }
        atomicAdd(&feat[img * 64 + oc], (p0 + p1) * (1.0f / 256.0f));
    }
}

// per image: fc 64->196 on pooled-mean features, concat audio -> write fwd
// stream and (reversed-l) bwd stream, both fp32 [1024][256]
__global__ __launch_bounds__(256)
void featfc_kernel(const float* __restrict__ feat,   // [1024][64]
                   const float* __restrict__ fcw,    // [196][64]
                   const float* __restrict__ fcb,    // [196]
                   const float* __restrict__ audio,  // [8][128*60]
                   float* __restrict__ xf, float* __restrict__ xb) {
    int img = blockIdx.x;
    int t = threadIdx.x;
    __shared__ float fl[64];
    if (t < 64) fl[t] = feat[img * 64 + t];
    __syncthreads();
    int b = img >> 7, l = img & 127;
    float v;
    if (t < VFEAT) {
        v = fcb[t];
#pragma unroll 8
        for (int k = 0; k < 64; ++k) v += fl[k] * fcw[t * 64 + k];
    } else {
        v = audio[(size_t)b * (LSEQ * NMFCC) + l * NMFCC + (t - VFEAT)];
    }
    xf[((size_t)(b * LSEQ + l) * DMODEL) + t] = v;
    xb[((size_t)(b * LSEQ + (LSEQ - 1 - l)) * DMODEL) + t] = v;
}

// ---------------------------------------------------------------------------
// Mamba building blocks
// ---------------------------------------------------------------------------
__global__ __launch_bounds__(256)
void rmsnorm_kernel(const float* __restrict__ x, const float* __restrict__ w,
                    unsigned short* __restrict__ out) {
    int row = blockIdx.x;
    int t = threadIdx.x;                       // 256 == DMODEL
    __shared__ float red[256];
    float v = x[(size_t)row * DMODEL + t];
    red[t] = v * v;
    __syncthreads();
    for (int s = 128; s > 0; s >>= 1) {
        if (t < s) red[t] += red[t + s];
        __syncthreads();
    }
    float rms = rsqrtf(red[0] * (1.0f / DMODEL) + 1e-5f);
    out[(size_t)row * DMODEL + t] = f32_to_bf16(v * rms * w[t]);
}

// Generic bf16 WMMA GEMM: C[M,N] = A[M,K] * W[N,K]^T (+ optional f32 residual)
template <int NT>
__global__ __launch_bounds__(256)
void gemm_kernel(const unsigned short* __restrict__ A,
                 const unsigned short* __restrict__ W,
                 const float* __restrict__ resid,
                 float* __restrict__ C, int M, int N, int K) {
    int wave = blockIdx.x * 8 + (threadIdx.x >> 5);
    int lane = threadIdx.x & 31;
    int tilesN = N / (16 * NT);
    int mtiles = M >> 4;
    if (wave >= mtiles * tilesN) return;
    int mt = wave / tilesN;
    int nt0 = (wave % tilesN) * NT;

    v8f acc[NT];
#pragma unroll
    for (int t = 0; t < NT; ++t) acc[t] = v8f{0, 0, 0, 0, 0, 0, 0, 0};

    const unsigned short* arow =
        A + ((size_t)(mt * 16 + (lane & 15))) * K + (lane >> 4) * 8;

    for (int k0 = 0; k0 < K; k0 += 32) {
        __builtin_prefetch(arow + k0 + 128, 0, 1);
        v16bf a = frag_load_a(arow + k0);
#pragma unroll
        for (int t = 0; t < NT; ++t) {
            const unsigned short* wrow =
                W + ((size_t)((nt0 + t) * 16 + (lane & 15))) * K +
                (lane >> 4) * 8 + k0;
            v16bf b = frag_load_b(wrow);
            acc[t] = wmma_bf16(a, b, acc[t]);
        }
    }

    int rbase = mt * 16 + 8 * (lane >> 4);
    int col0 = lane & 15;
#pragma unroll
    for (int t = 0; t < NT; ++t) {
        int c = (nt0 + t) * 16 + col0;
#pragma unroll
        for (int j = 0; j < 8; ++j) {
            size_t o = (size_t)(rbase + j) * N + c;
            float v = acc[t][j];
            if (resid) v += resid[o];
            C[o] = v;
        }
    }
}

// depthwise causal width-4 conv along l + SiLU; xz cols [0,512) are x-branch
__global__ __launch_bounds__(256)
void convsilu_kernel(const float* __restrict__ xz,    // [1024][1024]
                     const float* __restrict__ cw,    // [512][4]
                     const float* __restrict__ cb,    // [512]
                     float* __restrict__ xcf,         // [1024][512]
                     unsigned short* __restrict__ xcb) {
    int idx = blockIdx.x * 256 + threadIdx.x;
    if (idx >= NIMG * EDIM) return;
    int e = idx & (EDIM - 1);
    int row = idx >> 9;
    int b = row >> 7, l = row & 127;
    float s = cb[e];
#pragma unroll
    for (int k = 0; k < 4; ++k) {
        int l2 = l - 3 + k;
        if (l2 >= 0)
            s += xz[((size_t)(b * LSEQ + l2) * (2 * EDIM)) + e] * cw[e * 4 + k];
    }
    s = s / (1.0f + __expf(-s));                 // silu
    xcf[idx] = s;
    xcb[idx] = f32_to_bf16(s);
}

// delta = softplus(dt @ dt_w^T + dt_b); dt = dbc[:, :16]
__global__ __launch_bounds__(256)
void delta_kernel(const float* __restrict__ dbc,   // [1024][48]
                  const float* __restrict__ dtw,   // [512][16]
                  const float* __restrict__ dtb,   // [512]
                  float* __restrict__ delta) {     // [1024][512]
    int idx = blockIdx.x * 256 + threadIdx.x;
    if (idx >= NIMG * EDIM) return;
    int e = idx & (EDIM - 1);
    int row = idx >> 9;
    float s = dtb[e];
    const float* dt = dbc + (size_t)row * 48;
#pragma unroll
    for (int r = 0; r < DTR; ++r) s += dt[r] * dtw[e * DTR + r];
    delta[idx] = (s > 20.0f) ? s : log1pf(__expf(s));
}

// selective scan: one thread per (b, e), 16-state recurrence over L=128,
// fused with +D*xc, *silu(z), bf16 output for the out_proj GEMM
__global__ __launch_bounds__(256)
void scan_kernel(const float* __restrict__ delta,  // [1024][512]
                 const float* __restrict__ dbc,    // [1024][48] (B at 16, C at 32)
                 const float* __restrict__ xcf,    // [1024][512]
                 const float* __restrict__ xz,     // [1024][1024] (z at col 512)
                 const float* __restrict__ A_log,  // [512][16]
                 const float* __restrict__ Dp,     // [512]
                 unsigned short* __restrict__ ybf) {
    int idx = blockIdx.x * 256 + threadIdx.x;
    if (idx >= BSZ * EDIM) return;
    int e = idx & (EDIM - 1);
    int b = idx >> 9;

    float an[NST];
#pragma unroll
    for (int n = 0; n < NST; ++n) an[n] = -__expf(A_log[e * NST + n]);
    float Dv = Dp[e];
    float h[NST];
#pragma unroll
    for (int n = 0; n < NST; ++n) h[n] = 0.0f;

    for (int l = 0; l < LSEQ; ++l) {
        int row = b * LSEQ + l;
        float d  = delta[(size_t)row * EDIM + e];
        float xv = xcf[(size_t)row * EDIM + e];
        float zv = xz[(size_t)row * (2 * EDIM) + EDIM + e];
        const float* Bc = dbc + (size_t)row * 48;
        float y = 0.0f;
#pragma unroll
        for (int n = 0; n < NST; ++n) {
            float dA = __expf(d * an[n]);
            h[n] = dA * h[n] + d * Bc[16 + n] * xv;
            y += h[n] * Bc[32 + n];
        }
        y += Dv * xv;
        y *= zv / (1.0f + __expf(-zv));
        ybf[(size_t)row * EDIM + e] = f32_to_bf16(y);
    }
}

// head: x_last = [xf[:, L-1], xb_rev[:, 0]]; fc 512->256; fc2 256->2
__global__ __launch_bounds__(256)
void head_kernel(const float* __restrict__ xf, const float* __restrict__ xb,
                 const float* __restrict__ fcw, const float* __restrict__ fcb,
                 const float* __restrict__ fc2w, const float* __restrict__ fc2b,
                 float* __restrict__ out) {
    int b = blockIdx.x;
    int t = threadIdx.x;
    __shared__ float xl[512];
    __shared__ float h[256];
    xl[t]       = xf[((size_t)(b * LSEQ + LSEQ - 1) * DMODEL) + t];
    xl[256 + t] = xb[((size_t)(b * LSEQ + 0) * DMODEL) + t];
    __syncthreads();
    float s = fcb[t];
#pragma unroll 8
    for (int k = 0; k < 512; ++k) s += xl[k] * fcw[t * 512 + k];
    h[t] = s;
    __syncthreads();
    if (t < 2) {
        float o = fc2b[t];
        for (int k = 0; k < 256; ++k) o += h[k] * fc2w[t * 256 + k];
        out[b * 2 + t] = o;
    }
}

// ---------------------------------------------------------------------------
// Host-side orchestration
// ---------------------------------------------------------------------------
static inline int cdiv(long n, int d) { return (int)((n + d - 1) / d); }

extern "C" void kernel_launch(void* const* d_in, const int* in_sizes, int n_in,
                              void* d_out, int out_size, void* d_ws, size_t ws_size,
                              hipStream_t stream) {
    (void)in_sizes; (void)n_in; (void)out_size; (void)ws_size;

    const float* video = (const float*)d_in[0];
    const float* audio = (const float*)d_in[1];
    const float* w1    = (const float*)d_in[2];
    const float* b1    = (const float*)d_in[3];
    const float* w2    = (const float*)d_in[4];
    const float* b2    = (const float*)d_in[5];
    const float* fcw   = (const float*)d_in[6];
    const float* fcb   = (const float*)d_in[7];
    const float* fc_w  = (const float*)d_in[28];
    const float* fc_b  = (const float*)d_in[29];
    const float* fc2_w = (const float*)d_in[30];
    const float* fc2_b = (const float*)d_in[31];
    float* outp = (float*)d_out;

    // workspace carve (256B aligned)
    uint8_t* ws = (uint8_t*)d_ws;
    size_t off = 0;
    auto alloc = [&](size_t bytes) -> void* {
        off = (off + 255) & ~(size_t)255;
        void* p = ws + off;
        off += bytes;
        return p;
    };

    unsigned short* conv1out = (unsigned short*)alloc((size_t)NIMG * 32 * 32 * 32 * 2);
    unsigned short* w2pack   = (unsigned short*)alloc((size_t)9 * 64 * 32 * 2);
    float* feat = (float*)alloc((size_t)NIMG * 64 * 4);
    float* xf0  = (float*)alloc((size_t)NIMG * DMODEL * 4);
    float* xf1  = (float*)alloc((size_t)NIMG * DMODEL * 4);
    float* xb0  = (float*)alloc((size_t)NIMG * DMODEL * 4);
    float* xb1  = (float*)alloc((size_t)NIMG * DMODEL * 4);
    unsigned short* xn  = (unsigned short*)alloc((size_t)NIMG * DMODEL * 2);
    float* xz   = (float*)alloc((size_t)NIMG * 2 * EDIM * 4);
    float* xcf  = (float*)alloc((size_t)NIMG * EDIM * 4);
    unsigned short* xcb = (unsigned short*)alloc((size_t)NIMG * EDIM * 2);
    float* dbc  = (float*)alloc((size_t)NIMG * 48 * 4);
    float* dlt  = (float*)alloc((size_t)NIMG * EDIM * 4);
    unsigned short* ybf = (unsigned short*)alloc((size_t)NIMG * EDIM * 2);
    unsigned short* win_bf[2], *wxp_bf[2], *wout_bf[2];
    for (int d = 0; d < 2; ++d) {
        win_bf[d]  = (unsigned short*)alloc((size_t)NLAYER * 2 * EDIM * DMODEL * 2);
        wxp_bf[d]  = (unsigned short*)alloc((size_t)NLAYER * 48 * EDIM * 2);
        wout_bf[d] = (unsigned short*)alloc((size_t)NLAYER * DMODEL * EDIM * 2);
    }

    // weight conversions to bf16
    for (int d = 0; d < 2; ++d) {
        int base = 8 + 10 * d;
        int nin  = NLAYER * 2 * EDIM * DMODEL;
        int nxp  = NLAYER * 48 * EDIM;
        int nout = NLAYER * DMODEL * EDIM;
        convert_bf16_kernel<<<cdiv(nin, 256), 256, 0, stream>>>((const float*)d_in[base + 1], win_bf[d], nin);
        convert_bf16_kernel<<<cdiv(nxp, 256), 256, 0, stream>>>((const float*)d_in[base + 4], wxp_bf[d], nxp);
        convert_bf16_kernel<<<cdiv(nout, 256), 256, 0, stream>>>((const float*)d_in[base + 9], wout_bf[d], nout);
    }

    // ---- CNN front-end ----
    conv1_kernel<<<(NIMG * 32 * 32 * 32) / 256, 256, 0, stream>>>(video, w1, b1, conv1out);
    packw2_kernel<<<cdiv(9 * 64 * 32, 256), 256, 0, stream>>>(w2, w2pack);
    zero_f32_kernel<<<cdiv(NIMG * 64, 256), 256, 0, stream>>>(feat, NIMG * 64);
    conv2_wmma_kernel<<<(NIMG * 64) / 8, 256, 0, stream>>>(conv1out, w2pack, b2, feat);
    featfc_kernel<<<NIMG, 256, 0, stream>>>(feat, fcw, fcb, audio, xf0, xb0);

    // ---- bidirectional Mamba ----
    float* finals[2];
    for (int d = 0; d < 2; ++d) {
        int base = 8 + 10 * d;
        const float* normw = (const float*)d_in[base + 0];
        const float* convw = (const float*)d_in[base + 2];
        const float* convb = (const float*)d_in[base + 3];
        const float* dtw   = (const float*)d_in[base + 5];
        const float* dtb   = (const float*)d_in[base + 6];
        const float* Alog  = (const float*)d_in[base + 7];
        const float* Dp    = (const float*)d_in[base + 8];

        float* cur = d ? xb0 : xf0;
        float* nxt = d ? xb1 : xf1;
        for (int l = 0; l < NLAYER; ++l) {
            rmsnorm_kernel<<<NIMG, 256, 0, stream>>>(cur, normw + (size_t)l * DMODEL, xn);
            // in_proj: [1024,1024] = xn[1024,256] x W[1024,256]^T
            gemm_kernel<4><<<128, 256, 0, stream>>>(
                xn, win_bf[d] + (size_t)l * 2 * EDIM * DMODEL, nullptr, xz,
                NIMG, 2 * EDIM, DMODEL);
            convsilu_kernel<<<cdiv(NIMG * EDIM, 256), 256, 0, stream>>>(
                xz, convw + (size_t)l * EDIM * 4, convb + (size_t)l * EDIM, xcf, xcb);
            // x_proj: [1024,48] = xc[1024,512] x W[48,512]^T
            gemm_kernel<1><<<24, 256, 0, stream>>>(
                xcb, wxp_bf[d] + (size_t)l * 48 * EDIM, nullptr, dbc,
                NIMG, 48, EDIM);
            delta_kernel<<<cdiv(NIMG * EDIM, 256), 256, 0, stream>>>(
                dbc, dtw + (size_t)l * EDIM * DTR, dtb + (size_t)l * EDIM, dlt);
            scan_kernel<<<cdiv(BSZ * EDIM, 256), 256, 0, stream>>>(
                dlt, dbc, xcf, xz, Alog + (size_t)l * EDIM * NST,
                Dp + (size_t)l * EDIM, ybf);
            // out_proj + residual: [1024,256] = y[1024,512] x W[256,512]^T + cur
            gemm_kernel<4><<<32, 256, 0, stream>>>(
                ybf, wout_bf[d] + (size_t)l * DMODEL * EDIM, cur, nxt,
                NIMG, DMODEL, EDIM);
            float* tmp = cur; cur = nxt; nxt = tmp;
        }
        finals[d] = cur;
    }

    head_kernel<<<BSZ, 256, 0, stream>>>(finals[0], finals[1], fc_w, fc_b,
                                         fc2_w, fc2_b, outp);
}
